// BilinearAttention_48326972014791
// MI455X (gfx1250) — matrix operand order repaired and checked by reading, not compiled, not run on hardware
//
#include <hip/hip_runtime.h>
#include <hip/hip_bf16.h>

// Shapes (fixed by reference): B=4, Lq=Lk=256, Dq=Dk=Dv=H=512.
// d_out = [ h : 4*256*512 | attn : 4*256*256 | norm_attn : 4*256*256 ] fp32
// d_ws  = [ wq f32 | uh f32 | q_bf | k_bf | WqT_bf | WkT_bf | vT_bf | P_bf ] (~8.9 MB)

typedef __bf16        v16bf  __attribute__((ext_vector_type(16)));
typedef float         v8f    __attribute__((ext_vector_type(8)));
typedef unsigned int  u32x4  __attribute__((ext_vector_type(4)));
typedef int           async_v4i __attribute__((vector_size(16)));   // matches builtin proto

// ---- CDNA5 async global->LDS path (guarded probes) --------------------------
#if defined(__HIP_DEVICE_COMPILE__) && __has_builtin(__builtin_amdgcn_global_load_async_to_lds_b128)
#define HAVE_ASYNC 1
#else
#define HAVE_ASYNC 0
#endif

__device__ __forceinline__ void async_copy_b128(const __bf16* g, __bf16* l) {
#if HAVE_ASYNC
  __builtin_amdgcn_global_load_async_to_lds_b128(
      (__attribute__((address_space(1))) async_v4i*)(async_v4i*)g,
      (__attribute__((address_space(3))) async_v4i*)(async_v4i*)l,
      /*offset=*/0, /*cpol=*/0);
#else
  *(u32x4*)l = *(const u32x4*)g;   // synchronous staging fallback
#endif
}

template <int N>
__device__ __forceinline__ void wait_asynccnt() {
#if HAVE_ASYNC
#if __has_builtin(__builtin_amdgcn_s_wait_asynccnt)
  __builtin_amdgcn_s_wait_asynccnt(N);
#else
  if constexpr (N == 0) asm volatile("s_wait_asynccnt 0x0" ::: "memory");
  else                  asm volatile("s_wait_asynccnt 0x2" ::: "memory");
#endif
#endif
}

// ---- WMMA fragment gathers: each = two contiguous 16B LDS reads -------------
constexpr int LDS_STRIDE = 40;   // bf16 elems; 80 B row pitch, 16B-aligned chunks

union Frag { u32x4 q[2]; v16bf f; };

// A 16x32 bf16 (§7.12.2): lane m=L&15; K = {khi..khi+7, 16+khi..16+khi+7}, khi = (L<16)?0:8
__device__ __forceinline__ v16bf load_a_frag(const __bf16* As, int lane, int m_sub) {
  Frag u;
  const __bf16* base = As + (m_sub * 16 + (lane & 15)) * LDS_STRIDE;
  const int khi = (lane < 16) ? 0 : 8;
  u.q[0] = *(const u32x4*)(base + khi);
  u.q[1] = *(const u32x4*)(base + 16 + khi);
  return u.f;
}

// B 32x16 bf16: lane n=L&15; K = {khi..khi+7, 16+khi..16+khi+7}, khi = (L<16)?0:8.
// Bs is stored N-major (transposed weights), so this is two 16-byte chunks.
__device__ __forceinline__ v16bf load_b_frag(const __bf16* Bs, int lane, int n_sub) {
  Frag u;
  const __bf16* base = Bs + (n_sub * 16 + (lane & 15)) * LDS_STRIDE + ((lane < 16) ? 0 : 8);
  u.q[0] = *(const u32x4*)(base);
  u.q[1] = *(const u32x4*)(base + 16);
  return u.f;
}

// ---- 64x64 block-tile WMMA GEMM, double-buffered async LDS staging ----------
// A: [M][KDIM] bf16 row-major.  B: [N][KDIM] bf16 row-major (i.e. transposed).
// Out: fp32, leading dim ldOut. 8 waves: wave w -> rows m_sub=w>>1, cols 2*(w&1)..+1.
template <int LDA, int LDB, int KDIM>
__device__ __forceinline__ void wmma_gemm_64x64(
    const __bf16* __restrict__ Abase, const __bf16* __restrict__ Bbase,
    int m0, int n0, float* __restrict__ Out, int ldOut,
    const float* __restrict__ bias)
{
  __shared__ __bf16 As[2][64 * LDS_STRIDE];
  __shared__ __bf16 Bs[2][64 * LDS_STRIDE];

  const int tid = threadIdx.x, lane = tid & 31, wv = tid >> 5;
  const int m_sub  = wv >> 1;
  const int n_sub0 = (wv & 1) * 2;

  // staging map: thread t copies one 16B chunk of A and one of B per tile
  const int sr = tid >> 2;          // 0..63 (row of A tile / row of B^T tile)
  const int sc = (tid & 3) * 8;     // 0,8,16,24 bf16 elems

  auto issue = [&](int buf, int k0) {
    async_copy_b128(Abase + (size_t)(m0 + sr) * LDA + k0 + sc, &As[buf][sr * LDS_STRIDE + sc]);
    async_copy_b128(Bbase + (size_t)(n0 + sr) * LDB + k0 + sc, &Bs[buf][sr * LDS_STRIDE + sc]);
  };

  v8f acc0 = {}; v8f acc1 = {};
  constexpr int NK = KDIM / 32;
  int cur = 0;
  issue(0, 0);
  for (int it = 0; it < NK; ++it) {
    if (it + 1 < NK) {
      issue(cur ^ 1, (it + 1) * 32);
      wait_asynccnt<2>();           // per-wave in-order: tile 'it' (first 2 ops) done
    } else {
      wait_asynccnt<0>();
    }
    __syncthreads();                // all waves' chunks of tile 'it' visible

    const v16bf a  = load_a_frag(&As[cur][0], lane, m_sub);
    const v16bf b0 = load_b_frag(&Bs[cur][0], lane, n_sub0);
    const v16bf b1 = load_b_frag(&Bs[cur][0], lane, n_sub0 + 1);
    acc0 = __builtin_amdgcn_wmma_f32_16x16x32_bf16(false, a, false, b0, (short)0, acc0, false, false);
    acc1 = __builtin_amdgcn_wmma_f32_16x16x32_bf16(false, a, false, b1, (short)0, acc1, false, false);

    __syncthreads();                // everyone done reading 'cur' before it is re-filled
    cur ^= 1;
  }

  // C/D layout: VGPR r -> M = r + (lane<16 ? 0 : 8), N = lane&15.
  const int row_base = m0 + m_sub * 16 + ((lane < 16) ? 0 : 8);
  const int col0 = n0 + n_sub0 * 16 + (lane & 15);
  const int col1 = col0 + 16;
  const float bias0 = bias ? bias[col0] : 0.0f;
  const float bias1 = bias ? bias[col1] : 0.0f;
#pragma unroll
  for (int r = 0; r < 8; ++r) {
    Out[(size_t)(row_base + r) * ldOut + col0] = acc0[r] + bias0;
    Out[(size_t)(row_base + r) * ldOut + col1] = acc1[r] + bias1;
  }
}

// ---- Kernel 0: fp32 -> bf16 conversions + weight/value transposes -----------
__global__ __launch_bounds__(256) void convert_kernel(
    const float* __restrict__ query, const float* __restrict__ key,
    const float* __restrict__ Wq, const float* __restrict__ Wk,
    const float* __restrict__ value,
    __bf16* __restrict__ q_bf, __bf16* __restrict__ k_bf,
    __bf16* __restrict__ WqT, __bf16* __restrict__ WkT,
    __bf16* __restrict__ vT)
{
  const int i = blockIdx.x * 256 + threadIdx.x;           // grid covers 524288
  q_bf[i] = (__bf16)query[i];
  k_bf[i] = (__bf16)key[i];
  {
    // value[b][k][n] -> vT[b][n][k]   (coalesced read, strided write)
    const int kg = i >> 9, n = i & 511;                   // kg = b*256 + k
    const int b = kg >> 8, k = kg & 255;
    vT[((size_t)b * 512 + n) * 256 + k] = (__bf16)value[i];
  }
  if (i < 262144) {
    // W[k][n] -> WT[n][k]
    const int k = i >> 9, n = i & 511;
    WqT[(size_t)n * 512 + k] = (__bf16)Wq[i];
    WkT[(size_t)n * 512 + k] = (__bf16)Wk[i];
  }
}

// ---- Kernel 1: wq = query@Wq + bq ; uh = key@Wk  (M=1024,N=512,K=512) -------
__global__ __launch_bounds__(256) void proj_gemm_kernel(
    const __bf16* __restrict__ q_bf, const __bf16* __restrict__ k_bf,
    const __bf16* __restrict__ WqT,  const __bf16* __restrict__ WkT,
    const float* __restrict__ bq,
    float* __restrict__ wq_out, float* __restrict__ uh_out)
{
  const bool is_q = (blockIdx.z == 0);
  wmma_gemm_64x64<512, 512, 512>(is_q ? q_bf : k_bf, is_q ? WqT : WkT,
                                 blockIdx.y * 64, blockIdx.x * 64,
                                 is_q ? wq_out : uh_out, 512,
                                 is_q ? bq : nullptr);
}

// ---- Kernel 2: fused tanh-score + softmax (one block per (b,q)) -------------
__global__ __launch_bounds__(256) void score_softmax_kernel(
    const float* __restrict__ wq, const float* __restrict__ uh,
    const float* __restrict__ v,
    float* __restrict__ attn_out, float* __restrict__ nattn_out,
    __bf16* __restrict__ p_bf)
{
  const int H = 512, Lk = 256;
  const int bq_row = blockIdx.x;          // b*Lq + q
  const int b = bq_row >> 8;
  const int tid = threadIdx.x, lane = tid & 31, wv = tid >> 5;

  __shared__ float s_wq[512];
  __shared__ float s_v[512];
  __shared__ float s_scores[256];
  __shared__ float s_red[8];

  s_wq[tid]       = wq[(size_t)bq_row * H + tid];
  s_wq[tid + 256] = wq[(size_t)bq_row * H + tid + 256];
  s_v[tid]        = v[tid];
  s_v[tid + 256]  = v[tid + 256];
  __syncthreads();

  // wave wv sweeps key columns k = wv, wv+8, ...; float4 over H, 32-lane reduce
  for (int k = wv; k < Lk; k += 8) {
    const float* __restrict__ uh_row = uh + (size_t)(b * Lk + k) * H;
    float acc = 0.0f;
#pragma unroll
    for (int i = 0; i < 4; ++i) {
      const int h = i * 128 + lane * 4;
      const float4 u  = *(const float4*)(uh_row + h);
      const float4 w4 = *(const float4*)(s_wq + h);
      const float4 v4 = *(const float4*)(s_v + h);
      acc += v4.x * tanhf(w4.x + u.x);
      acc += v4.y * tanhf(w4.y + u.y);
      acc += v4.z * tanhf(w4.z + u.z);
      acc += v4.w * tanhf(w4.w + u.w);
    }
#pragma unroll
    for (int off = 16; off > 0; off >>= 1)
      acc += __shfl_xor(acc, off, 32);
    if (lane == 0) {
      s_scores[k] = acc;
      attn_out[(size_t)bq_row * Lk + k] = acc;   // raw scores
    }
  }
  __syncthreads();

  // block softmax over 256 scores (thread t owns score t)
  const float val = s_scores[tid];
  float m = val;
#pragma unroll
  for (int off = 16; off > 0; off >>= 1)
    m = fmaxf(m, __shfl_xor(m, off, 32));
  if (lane == 0) s_red[wv] = m;
  __syncthreads();
  if (tid == 0) {
    float mm = s_red[0];
#pragma unroll
    for (int i = 1; i < 8; ++i) mm = fmaxf(mm, s_red[i]);
    s_red[0] = mm;
  }
  __syncthreads();
  m = s_red[0];
  const float e = __expf(val - m);
  float s = e;
#pragma unroll
  for (int off = 16; off > 0; off >>= 1)
    s += __shfl_xor(s, off, 32);
  __syncthreads();
  if (lane == 0) s_red[wv] = s;
  __syncthreads();
  if (tid == 0) {
    float ss = 0.0f;
#pragma unroll
    for (int i = 0; i < 8; ++i) ss += s_red[i];
    s_red[0] = ss;
  }
  __syncthreads();
  const float p = e * (1.0f / s_red[0]);
  nattn_out[(size_t)bq_row * Lk + tid] = p;
  p_bf[(size_t)bq_row * Lk + tid] = (__bf16)p;
}

// ---- Kernel 3: h = norm_attn @ value, per batch (M=256,N=512,K=256) ---------
__global__ __launch_bounds__(256) void av_gemm_kernel(
    const __bf16* __restrict__ p_bf, const __bf16* __restrict__ vT,
    float* __restrict__ h_out)
{
  const int b = blockIdx.z;
  wmma_gemm_64x64<256, 256, 256>(p_bf + (size_t)b * 256 * 256,
                                 vT   + (size_t)b * 512 * 256,
                                 blockIdx.y * 64, blockIdx.x * 64,
                                 h_out + (size_t)b * 256 * 512, 512, nullptr);
}

// ---------------------------------------------------------------------------
extern "C" void kernel_launch(void* const* d_in, const int* in_sizes, int n_in,
                              void* d_out, int out_size, void* d_ws, size_t ws_size,
                              hipStream_t stream) {
  const float* query = (const float*)d_in[0];
  const float* key   = (const float*)d_in[1];
  const float* value = (const float*)d_in[2];
  const float* Wq    = (const float*)d_in[3];
  const float* bq    = (const float*)d_in[4];
  const float* Wk    = (const float*)d_in[5];
  const float* v     = (const float*)d_in[6];

  float* out       = (float*)d_out;
  float* h_out     = out;                              // 4*256*512
  float* attn_out  = out + 4 * 256 * 512;              // 4*256*256
  float* nattn_out = attn_out + 4 * 256 * 256;

  float*  wq_ws = (float*)d_ws;                        // 524288 f32
  float*  uh_ws = wq_ws + 524288;                      // 524288 f32
  __bf16* q_bf  = (__bf16*)(uh_ws + 524288);           // 524288 bf16
  __bf16* k_bf  = q_bf + 524288;                       // 524288
  __bf16* WqT   = k_bf + 524288;                       // 262144
  __bf16* WkT   = WqT + 262144;                        // 262144
  __bf16* vT    = WkT + 262144;                        // 524288
  __bf16* p_bf  = vT + 524288;                         // 262144

  const dim3 blk(256);
  convert_kernel<<<dim3(2048), blk, 0, stream>>>(query, key, Wq, Wk, value,
                                                 q_bf, k_bf, WqT, WkT, vT);
  proj_gemm_kernel<<<dim3(8, 16, 2), blk, 0, stream>>>(q_bf, k_bf, WqT, WkT, bq,
                                                       wq_ws, uh_ws);
  score_softmax_kernel<<<dim3(4 * 256), blk, 0, stream>>>(wq_ws, uh_ws, v,
                                                          attn_out, nattn_out, p_bf);
  av_gemm_kernel<<<dim3(8, 4, 4), blk, 0, stream>>>(p_bf, vT, h_out);
}
